// ContPeepholeLSTMFunc_Delay_23450521436409
// MI455X (gfx1250) — matrix operand order repaired and checked by reading, not compiled, not run on hardware
//
#include <hip/hip_runtime.h>
#include <hip/hip_bf16.h>

// ---------------------------------------------------------------------------
// Continuous peephole LSTM derivative step for MI455X (gfx1250).
// All GEMM work mapped onto v_wmma_f32_16x16x32_bf16 (f32 accumulate).
// B=32768, I=256, H=512.  ~275 GFLOP of GEMM -> matrix-core bound.
// ---------------------------------------------------------------------------

typedef __attribute__((ext_vector_type(16))) __bf16 v16bf;
typedef __attribute__((ext_vector_type(8)))  __bf16 v8bf;
typedef __attribute__((ext_vector_type(8)))  float  v8f;

#define B_DIM 32768
#define I_DIM 256
#define H_DIM 512

#define WMMA_BF16(A, Bm, C) \
  __builtin_amdgcn_wmma_f32_16x16x32_bf16(false, (A), false, (Bm), (short)0, (C), false, false)

static __device__ __forceinline__ float sigmoid_f(float x) {
  return 1.0f / (1.0f + __expf(-x));
}

// A fragment (16x32, bf16).  ISA layout: lanes 0-15 hold M=lane, lanes 16-31
// hold the same M but K offset by 8/16; per lane the 16 elements are two
// contiguous K-runs of 8: [hi*8 .. hi*8+7] and [16+hi*8 .. 16+hi*8+7].
// `p` points at row start + k0 + hi*8 in an f32 row-major source.
static __device__ __forceinline__ v16bf load_a_f32(const float* p) {
  float4 a0 = *(const float4*)(p);
  float4 a1 = *(const float4*)(p + 4);
  float4 b0 = *(const float4*)(p + 16);
  float4 b1 = *(const float4*)(p + 20);
  v16bf r;
  r[0]  = (__bf16)a0.x; r[1]  = (__bf16)a0.y; r[2]  = (__bf16)a0.z; r[3]  = (__bf16)a0.w;
  r[4]  = (__bf16)a1.x; r[5]  = (__bf16)a1.y; r[6]  = (__bf16)a1.z; r[7]  = (__bf16)a1.w;
  r[8]  = (__bf16)b0.x; r[9]  = (__bf16)b0.y; r[10] = (__bf16)b0.z; r[11] = (__bf16)b0.w;
  r[12] = (__bf16)b1.x; r[13] = (__bf16)b1.y; r[14] = (__bf16)b1.z; r[15] = (__bf16)b1.w;
  return r;
}

// Same fragment from an already-bf16 row-major source.
static __device__ __forceinline__ v16bf load_a_bf16(const __bf16* p) {
  v8bf a = *(const v8bf*)(p);
  v8bf b = *(const v8bf*)(p + 16);
  v16bf r;
#pragma unroll
  for (int i = 0; i < 8; ++i) { r[i] = a[i]; r[i + 8] = b[i]; }
  return r;
}

// B fragment (32x16, bf16): lane L holds column N = L%16, K = hi*8+i and 16+hi*8+i.
// `p` = Bt + (k0+hi*8)*H + n0 + (lane&15), Bt is K-major (pre-transposed weight).
static __device__ __forceinline__ v16bf load_b_bf16(const __bf16* p) {
  v16bf r;
#pragma unroll
  for (int i = 0; i < 8; ++i) { r[i] = p[(size_t)i * H_DIM]; r[i + 8] = p[(size_t)(16 + i) * H_DIM]; }
  return r;
}

// ---------------------------------------------------------------------------
// Weight prep: dst[k*H + n] = (bf16) src[n*K + k]   (src is H x K row-major)
// ---------------------------------------------------------------------------
__global__ void k_tcvt(const float* __restrict__ src, __bf16* __restrict__ dst, int K) {
  int id = blockIdx.x * 256 + threadIdx.x;
  int k = id / H_DIM;
  int n = id - k * H_DIM;
  dst[id] = (__bf16)src[(size_t)n * K + k];
}

__global__ void k_tcvt_add(const float* __restrict__ u, const float* __restrict__ p,
                           __bf16* __restrict__ dst) {
  int id = blockIdx.x * 256 + threadIdx.x;
  int k = id >> 9;
  int n = id & (H_DIM - 1);
  size_t s = (size_t)n * H_DIM + k;
  dst[id] = (__bf16)(u[s] + p[s]);
}

// ---------------------------------------------------------------------------
// Phase A: all GEMMs except the Po peephole, plus the gate nonlinearity chain.
// Block = 64 batch rows x 64 hidden cols, 8 waves; each wave: 1 m-tile (16) x
// 2 n-tiles (16), carrying 8 f32 accumulators per output element group.
// ---------------------------------------------------------------------------
__global__ __launch_bounds__(256)
void phaseA(const float* __restrict__ x,      const float* __restrict__ h_past,
            const float* __restrict__ c_past, const float* __restrict__ dh,
            const float* __restrict__ dcp,    const float* __restrict__ cg,
            const __bf16* __restrict__ WtI, const __bf16* __restrict__ WtF,
            const __bf16* __restrict__ WtO, const __bf16* __restrict__ WtC,
            const __bf16* __restrict__ UtI, const __bf16* __restrict__ UtF,
            const __bf16* __restrict__ UtO, const __bf16* __restrict__ UtC,
            const __bf16* __restrict__ PtI, const __bf16* __restrict__ PtF,
            const __bf16* __restrict__ UptI, const __bf16* __restrict__ UptF,
            const float* __restrict__ bUi, const float* __restrict__ bUf,
            const float* __restrict__ bUo, const float* __restrict__ bUc,
            const float* __restrict__ bPi, const float* __restrict__ bPf,
            __bf16* __restrict__ c16, __bf16* __restrict__ dc16,
            float* __restrict__ s_o, float* __restrict__ s_dC)
{
  const int lane  = threadIdx.x & 31;
  const int wave  = threadIdx.x >> 5;
  const int m16   = wave & 3;      // 4 m-tiles of 16 -> 64 rows
  const int nsel  = wave >> 2;     // 2 groups of 2 n-tiles -> 64 cols
  const int b0    = blockIdx.x * 64;
  const int n0    = blockIdx.y * 64;
  const int mlane = lane & 15;
  const int hi    = lane >> 4;
  const int arow  = b0 + m16 * 16 + mlane;

  v8f z = {0.f, 0.f, 0.f, 0.f, 0.f, 0.f, 0.f, 0.f};
  v8f ai[2], af[2], ac[2], so[2], dA[2], dB[2], dD[2], dC[2];
#pragma unroll
  for (int t = 0; t < 2; ++t) {
    ai[t] = z; af[t] = z; ac[t] = z; so[t] = z;
    dA[t] = z; dB[t] = z; dD[t] = z; dC[t] = z;
  }

  // ---- K over input dim (I = 256): x@W*, control_grad@W* --------------------
  const float* xrow  = x  + (size_t)arow * I_DIM + hi * 8;
  const float* cgrow = cg + (size_t)arow * I_DIM + hi * 8;
  for (int k0 = 0; k0 < I_DIM; k0 += 32) {
    v16bf Ax  = load_a_f32(xrow + k0);
    v16bf Acg = load_a_f32(cgrow + k0);
    const size_t brow = (size_t)(k0 + hi * 8) * H_DIM + mlane;
#pragma unroll
    for (int t = 0; t < 2; ++t) {
      const int nb = n0 + (nsel * 2 + t) * 16;
      v16bf Bi = load_b_bf16(WtI + brow + nb);
      ai[t] = WMMA_BF16(Ax,  Bi, ai[t]);
      dA[t] = WMMA_BF16(Acg, Bi, dA[t]);
      v16bf Bf = load_b_bf16(WtF + brow + nb);
      af[t] = WMMA_BF16(Ax,  Bf, af[t]);
      dB[t] = WMMA_BF16(Acg, Bf, dB[t]);
      v16bf Bc = load_b_bf16(WtC + brow + nb);
      ac[t] = WMMA_BF16(Ax,  Bc, ac[t]);
      dD[t] = WMMA_BF16(Acg, Bc, dD[t]);
      v16bf Bo = load_b_bf16(WtO + brow + nb);
      so[t] = WMMA_BF16(Ax,  Bo, so[t]);
      dC[t] = WMMA_BF16(Acg, Bo, dC[t]);
    }
  }

  // ---- K over hidden dim (H = 512): h@U*, c@P*, dh@(U+P)/U ------------------
  const float* hrow  = h_past + (size_t)arow * H_DIM + hi * 8;
  const float* crow  = c_past + (size_t)arow * H_DIM + hi * 8;
  const float* dhrow = dh     + (size_t)arow * H_DIM + hi * 8;
  for (int k0 = 0; k0 < H_DIM; k0 += 32) {
    v16bf Ah  = load_a_f32(hrow + k0);
    v16bf Ac  = load_a_f32(crow + k0);
    v16bf Adh = load_a_f32(dhrow + k0);
    const size_t brow = (size_t)(k0 + hi * 8) * H_DIM + mlane;
#pragma unroll
    for (int t = 0; t < 2; ++t) {
      const int nb = n0 + (nsel * 2 + t) * 16;
      v16bf Bui = load_b_bf16(UtI + brow + nb);
      ai[t] = WMMA_BF16(Ah, Bui, ai[t]);
      v16bf Bpi = load_b_bf16(PtI + brow + nb);
      ai[t] = WMMA_BF16(Ac, Bpi, ai[t]);
      v16bf Bup = load_b_bf16(UptI + brow + nb);
      dA[t] = WMMA_BF16(Adh, Bup, dA[t]);
      v16bf Buf = load_b_bf16(UtF + brow + nb);
      af[t] = WMMA_BF16(Ah, Buf, af[t]);
      v16bf Bpf = load_b_bf16(PtF + brow + nb);
      af[t] = WMMA_BF16(Ac, Bpf, af[t]);
      v16bf Bpu = load_b_bf16(UptF + brow + nb);
      dB[t] = WMMA_BF16(Adh, Bpu, dB[t]);
      v16bf Buc = load_b_bf16(UtC + brow + nb);
      ac[t] = WMMA_BF16(Ah,  Buc, ac[t]);
      dD[t] = WMMA_BF16(Adh, Buc, dD[t]);
      v16bf Buo = load_b_bf16(UtO + brow + nb);
      so[t] = WMMA_BF16(Ah,  Buo, so[t]);
      dC[t] = WMMA_BF16(Adh, Buo, dC[t]);
    }
  }

  // ---- epilogue: gate nonlinearities + cell state + derivative chain --------
#pragma unroll
  for (int t = 0; t < 2; ++t) {
    const int n    = n0 + (nsel * 2 + t) * 16 + mlane;
    const float bi = bUi[n] + bPi[n];
    const float bf = bUf[n] + bPf[n];
    const float bc = bUc[n];
    const float bo = bUo[n];
#pragma unroll
    for (int j = 0; j < 8; ++j) {
      const int   brow_ = b0 + m16 * 16 + j + 8 * hi;   // C/D layout: M = j + 8*hi
      const size_t idx  = (size_t)brow_ * H_DIM + n;
      const float cp   = c_past[idx];
      const float dcpv = dcp[idx];
      const float it   = sigmoid_f(ai[t][j] + bi);
      const float ft   = sigmoid_f(af[t][j] + bf);
      const float ctl  = tanhf(ac[t][j] + bc);
      const float c_t  = ft * cp + it * ctl;
      const float didt = it * (1.f - it) * dA[t][j];
      const float dfdt = ft * (1.f - ft) * dB[t][j];
      const float dctl = (1.f - ctl * ctl) * dD[t][j];
      const float dc   = dfdt * cp + ft * dcpv + didt * ctl + it * dctl;
      c16[idx]  = (__bf16)c_t;
      dc16[idx] = (__bf16)dc;
      s_o[idx]  = so[t][j] + bo;      // x@Wo + h@Uo + bUo
      s_dC[idx] = dC[t][j];           // cg@Wo + dh@Uo
    }
  }
}

// ---------------------------------------------------------------------------
// Phase B: c_t@Po^T and dc_dt@Po^T, then o-gate and dh_dt output.
// ---------------------------------------------------------------------------
__global__ __launch_bounds__(256)
void phaseB(const __bf16* __restrict__ c16, const __bf16* __restrict__ dc16,
            const __bf16* __restrict__ PtO,
            const float* __restrict__ s_o, const float* __restrict__ s_dC,
            const float* __restrict__ bPo, float* __restrict__ out)
{
  const int lane  = threadIdx.x & 31;
  const int wave  = threadIdx.x >> 5;
  const int m16   = wave & 3;
  const int nsel  = wave >> 2;
  const int b0    = blockIdx.x * 64;
  const int n0    = blockIdx.y * 64;
  const int mlane = lane & 15;
  const int hi    = lane >> 4;
  const int arow  = b0 + m16 * 16 + mlane;

  v8f z = {0.f, 0.f, 0.f, 0.f, 0.f, 0.f, 0.f, 0.f};
  v8f t1[2] = {z, z}, t2[2] = {z, z};

  const __bf16* crow  = c16  + (size_t)arow * H_DIM + hi * 8;
  const __bf16* dcrow = dc16 + (size_t)arow * H_DIM + hi * 8;
  for (int k0 = 0; k0 < H_DIM; k0 += 32) {
    v16bf Ac  = load_a_bf16(crow + k0);
    v16bf Adc = load_a_bf16(dcrow + k0);
    const size_t brow = (size_t)(k0 + hi * 8) * H_DIM + mlane;
#pragma unroll
    for (int t = 0; t < 2; ++t) {
      const int nb = n0 + (nsel * 2 + t) * 16;
      v16bf Bpo = load_b_bf16(PtO + brow + nb);
      t1[t] = WMMA_BF16(Ac,  Bpo, t1[t]);
      t2[t] = WMMA_BF16(Adc, Bpo, t2[t]);
    }
  }

#pragma unroll
  for (int t = 0; t < 2; ++t) {
    const int n   = n0 + (nsel * 2 + t) * 16 + mlane;
    const float bp = bPo[n];
#pragma unroll
    for (int j = 0; j < 8; ++j) {
      const int   brow_ = b0 + m16 * 16 + j + 8 * hi;
      const size_t idx  = (size_t)brow_ * H_DIM + n;
      const float ot   = sigmoid_f(s_o[idx] + t1[t][j] + bp);
      const float dCv  = s_dC[idx] + t2[t][j];
      const float dodt = ot * (1.f - ot) * dCv;
      const float c_t  = (float)c16[idx];
      const float th   = tanhf(c_t);
      const float dc   = (float)dc16[idx];
      out[idx] = dodt * th + ot * (1.f - th * th) * dc;
    }
  }
}

// ---------------------------------------------------------------------------
// Host launcher.
// Workspace layout (bytes):
//   [0 .. 5.77MB)   bf16 transposed weights (4xW, 4xU, 3xP, 2x(U+P))
//   then c16, dc16  (B*H bf16 each, 33.5MB each)
//   then s_o, s_dC  (B*H f32 each,  67MB each)     total ~207MB
// ---------------------------------------------------------------------------
extern "C" void kernel_launch(void* const* d_in, const int* in_sizes, int n_in,
                              void* d_out, int out_size, void* d_ws, size_t ws_size,
                              hipStream_t stream) {
  (void)in_sizes; (void)n_in; (void)out_size; (void)ws_size;
  const float* x   = (const float*)d_in[0];
  const float* h   = (const float*)d_in[1];
  const float* c   = (const float*)d_in[2];
  const float* dh  = (const float*)d_in[3];
  const float* dcp = (const float*)d_in[4];
  const float* cg  = (const float*)d_in[5];
  const float* Wi  = (const float*)d_in[6];
  const float* Wf  = (const float*)d_in[7];
  const float* Wo  = (const float*)d_in[8];
  const float* Wc  = (const float*)d_in[9];
  const float* Ui  = (const float*)d_in[10];
  const float* Uf  = (const float*)d_in[11];
  const float* Uo  = (const float*)d_in[12];
  const float* Uc  = (const float*)d_in[13];
  const float* bUi = (const float*)d_in[14];
  const float* bUf = (const float*)d_in[15];
  const float* bUo = (const float*)d_in[16];
  const float* bUc = (const float*)d_in[17];
  const float* Pi  = (const float*)d_in[18];
  const float* Pf  = (const float*)d_in[19];
  const float* Po  = (const float*)d_in[20];
  const float* bPi = (const float*)d_in[21];
  const float* bPf = (const float*)d_in[22];
  const float* bPo = (const float*)d_in[23];

  const size_t IH = (size_t)I_DIM * H_DIM;   // 131072
  const size_t HH = (size_t)H_DIM * H_DIM;   // 262144
  const size_t BH = (size_t)B_DIM * H_DIM;   // 16777216

  __bf16* w16  = (__bf16*)d_ws;
  __bf16* WtI  = w16;            __bf16* WtF = WtI + IH;
  __bf16* WtO  = WtF + IH;       __bf16* WtC = WtO + IH;
  __bf16* UtI  = WtC + IH;       __bf16* UtF = UtI + HH;
  __bf16* UtO  = UtF + HH;       __bf16* UtC = UtO + HH;
  __bf16* PtI  = UtC + HH;       __bf16* PtF = PtI + HH;
  __bf16* PtO  = PtF + HH;
  __bf16* UptI = PtO + HH;       __bf16* UptF = UptI + HH;
  __bf16* c16  = UptF + HH;      __bf16* dc16 = c16 + BH;
  float*  fws  = (float*)(dc16 + BH);
  float*  s_o  = fws;            float* s_dC = fws + BH;

  const int blkW = (int)(IH / 256);   // 512
  const int blkU = (int)(HH / 256);   // 1024
  k_tcvt<<<blkW, 256, 0, stream>>>(Wi, WtI, I_DIM);
  k_tcvt<<<blkW, 256, 0, stream>>>(Wf, WtF, I_DIM);
  k_tcvt<<<blkW, 256, 0, stream>>>(Wo, WtO, I_DIM);
  k_tcvt<<<blkW, 256, 0, stream>>>(Wc, WtC, I_DIM);
  k_tcvt<<<blkU, 256, 0, stream>>>(Ui, UtI, H_DIM);
  k_tcvt<<<blkU, 256, 0, stream>>>(Uf, UtF, H_DIM);
  k_tcvt<<<blkU, 256, 0, stream>>>(Uo, UtO, H_DIM);
  k_tcvt<<<blkU, 256, 0, stream>>>(Uc, UtC, H_DIM);
  k_tcvt<<<blkU, 256, 0, stream>>>(Pi, PtI, H_DIM);
  k_tcvt<<<blkU, 256, 0, stream>>>(Pf, PtF, H_DIM);
  k_tcvt<<<blkU, 256, 0, stream>>>(Po, PtO, H_DIM);
  k_tcvt_add<<<blkU, 256, 0, stream>>>(Ui, Pi, UptI);
  k_tcvt_add<<<blkU, 256, 0, stream>>>(Uf, Pf, UptF);

  dim3 grid(B_DIM / 64, H_DIM / 64);   // 512 x 8
  phaseA<<<grid, 256, 0, stream>>>(x, h, c, dh, dcp, cg,
                                   WtI, WtF, WtO, WtC,
                                   UtI, UtF, UtO, UtC,
                                   PtI, PtF, UptI, UptF,
                                   bUi, bUf, bUo, bUc, bPi, bPf,
                                   c16, dc16, s_o, s_dC);
  phaseB<<<grid, 256, 0, stream>>>(c16, dc16, PtO, s_o, s_dC, bPo,
                                   (float*)d_out);
}
